// LoopedMLPForLM_2233382994583
// MI455X (gfx1250) — matrix operand order repaired and checked by reading, not compiled, not run on hardware
//
#include <hip/hip_runtime.h>
#include <hip/hip_bf16.h>
#include <math.h>

// ---------------------------------------------------------------------------
// LoopedMLPForLM on MI455X (gfx1250, wave32, WMMA).
//
// Precision: bf16 A/B operands, f32 WMMA accumulation (v_wmma_f32_16x16x32_bf16),
// f32 bias/tanh/x_proj epilogues. Weights converted to bf16 once (W_rec 2MB and
// lm_head 65MB live in the 192MB L2). Logits write (1.05 GB) is the HBM floor
// (~45us @ 23.3 TB/s).
//
// GEMM core v2: block = 8 waves = 256 threads, block tile 256(M) x 64(N).
//  - Per wave: 32(M) x 64(N) = 2 A-frags x 4 N-tiles = 8 WMMA accumulators.
//  - B tile (64 cols x 32 K bf16 = 4KB) staged in LDS once per block per
//    K-step, double-buffered (8KB LDS), one barrier per K-step. This cuts B
//    traffic 8x vs per-wave fetch: 64 WMMAs (1.05 MFLOP) per 20KB moved
//    -> ~52 FLOP/byte of L2 traffic, inside plausible L2 bandwidth at WMMA peak.
//  - LDS layout Bs[col][k] (64B row stride): a wave's 4 fragment reads give a
//    uniform 4-request-per-bank pattern (the 1KB/wave minimum), all 16B aligned
//    so they lower to ds_load_b128 / ds_store_b128.
// ---------------------------------------------------------------------------

typedef __attribute__((ext_vector_type(16))) __bf16 bf16x16;
typedef __attribute__((ext_vector_type(8)))  __bf16 bf16x8;
typedef __attribute__((ext_vector_type(8)))  float  f32x8;

#define HID      1024
#define VOCABSZ  32000
#define BATCH    4
#define SEQ      2048
#define M_TOTAL  (BATCH * SEQ)     // 8192 rows
#define EQ_STEPS 20

union AFrag { bf16x16 v; bf16x8 h8[2]; __bf16 e[16]; };

#define WMMA_BF16(A, B, C) \
  __builtin_amdgcn_wmma_f32_16x16x32_bf16(false, (A), false, (B), (short)0, (C), false, false)

// ------------------------- elementwise helpers -----------------------------

__global__ __launch_bounds__(256) void k_f32_to_bf16(
    const float* __restrict__ src, __bf16* __restrict__ dst, int n) {
  int i = blockIdx.x * blockDim.x + threadIdx.x;
  int stride = gridDim.x * blockDim.x;
  for (; i < n; i += stride) dst[i] = (__bf16)src[i];
}

__global__ __launch_bounds__(256) void k_zero_bf16(
    __bf16* __restrict__ dst, int n) {
  int i = blockIdx.x * blockDim.x + threadIdx.x;
  int stride = gridDim.x * blockDim.x;
  for (; i < n; i += stride) dst[i] = (__bf16)0.0f;
}

// ---------------------------------------------------------------------------
// (1) fused embedding gather + input projection (2% of FLOPs; simple per-wave
//     16x64 form, A built on the fly from token_emb[x] + pos_emb).
//     xproj = (tok_emb[x] + pos_emb) @ W_in^T + b_in
// ---------------------------------------------------------------------------
__global__ __launch_bounds__(256) void k_embed_inproj(
    const int*   __restrict__ tokens,    // [M_TOTAL]
    const float* __restrict__ tok_emb,   // [VOCAB, HID]
    const float* __restrict__ pos_emb,   // [SEQ, HID]
    const __bf16* __restrict__ Wb,       // [HID, HID] bf16 (row-major W_in)
    const float* __restrict__ bias,      // [HID]
    float*       __restrict__ xproj)     // [M_TOTAL, HID]
{
  const int lane    = threadIdx.x & 31;
  const int wave    = threadIdx.x >> 5;
  const int gw      = blockIdx.x * 8 + wave;
  const int ngroups = HID / 64;                 // 16
  const int m_base  = (gw / ngroups) * 16;
  const int n_base  = (gw % ngroups) * 64;

  const int arow = m_base + (lane & 15);
  const int tok  = tokens[arow];
  const int srow = arow & (SEQ - 1);
  const float* te = tok_emb + (size_t)tok  * HID;
  const float* pe = pos_emb + (size_t)srow * HID;
  const int koff    = (lane >> 4) << 3;         // 0 or 8
  const int kb_lane = (lane >> 4) << 4;         // 0 or 16
  const int ncol    = n_base + (lane & 15);

  f32x8 acc[4] = {};
  for (int k0 = 0; k0 < HID; k0 += 32) {
    AFrag a;
#pragma unroll
    for (int j = 0; j < 8; ++j) {
      a.e[j]     = (__bf16)(te[k0 + kb_lane + j]     + pe[k0 + kb_lane + j]);
      a.e[8 + j] = (__bf16)(te[k0 + kb_lane + 8 + j] + pe[k0 + kb_lane + 8 + j]);
    }
#pragma unroll
    for (int t = 0; t < 4; ++t) {
      bf16x16 b = *(const bf16x16*)(Wb + (size_t)(ncol + 16 * t) * HID + k0 + kb_lane);
      acc[t] = WMMA_BF16(a.v, b, acc[t]);
    }
  }
  const int rbase = m_base + ((lane >> 4) << 3);
#pragma unroll
  for (int t = 0; t < 4; ++t) {
    const int cg = n_base + 16 * t + (lane & 15);
    const float bv = bias[cg];
#pragma unroll
    for (int v = 0; v < 8; ++v)
      xproj[(size_t)(rbase + v) * HID + cg] = acc[t][v] + bv;
  }
}

// ---------------------------------------------------------------------------
// (2) recurrent step: h_out = tanh(xproj + h_in @ W_rec^T + b_rec)
//     Block tile 256(M) x 64(N), LDS-staged B, double buffered.
// ---------------------------------------------------------------------------
__global__ __launch_bounds__(256) void k_rnn_step(
    const __bf16* __restrict__ hin,      // [M_TOTAL, HID] bf16
    const __bf16* __restrict__ Wrec,     // [HID, HID] bf16
    const float*  __restrict__ xproj,    // [M_TOTAL, HID] f32
    const float*  __restrict__ bias,     // [HID]
    __bf16*       __restrict__ hout)     // [M_TOTAL, HID] bf16
{
  __shared__ __bf16 Bs[2][64 * 32];      // [buf][col*32 + k], 4KB per buffer

  const int tid     = threadIdx.x;
  const int lane    = tid & 31;
  const int wave    = tid >> 5;
  const int ngroups = HID / 64;                  // 16
  const int m_blk   = (blockIdx.x / ngroups) * 256;
  const int n_base  = (blockIdx.x % ngroups) * 64;

  // staging: thread tid copies 8 bf16 (16B): col = tid/4, k = (tid%4)*8
  const int scol  = tid >> 2;
  const int skoff = (tid & 3) * 8;
  const __bf16* wsrc = Wrec + (size_t)(n_base + scol) * HID + skoff;
  const int sidx = scol * 32 + skoff;

  const int m_wave   = m_blk + wave * 32;
  const __bf16* ap0  = hin + (size_t)(m_wave + (lane & 15)) * HID;
  const __bf16* ap1  = ap0 + (size_t)16 * HID;
  const int koff     = (lane >> 4) << 3;         // 0 or 8
  const int kb_lane  = (lane >> 4) << 4;         // 0 or 16
  const int bcolbase = (lane & 15) * 32 + kb_lane;

  f32x8 acc[2][4] = {};

  // preload K-chunk 0
  *(bf16x8*)(&Bs[0][sidx]) = *(const bf16x8*)(wsrc);
  __syncthreads();

  const int NK = HID / 32;                       // 32
  for (int kc = 0; kc < NK; ++kc) {
    const int cur = kc & 1;
    if (kc + 1 < NK)   // fill the other buffer while this one is consumed
      *(bf16x8*)(&Bs[cur ^ 1][sidx]) = *(const bf16x8*)(wsrc + (size_t)(kc + 1) * 32);

    const int k0 = kc * 32;
    AFrag a0, a1;
    a0.h8[0] = *(const bf16x8*)(ap0 + k0 + kb_lane);
    a0.h8[1] = *(const bf16x8*)(ap0 + k0 + kb_lane + 8);
    a1.h8[0] = *(const bf16x8*)(ap1 + k0 + kb_lane);
    a1.h8[1] = *(const bf16x8*)(ap1 + k0 + kb_lane + 8);
#pragma unroll
    for (int t = 0; t < 4; ++t) {
      bf16x16 b = *(const bf16x16*)(&Bs[cur][bcolbase + t * (16 * 32)]);
      acc[0][t] = WMMA_BF16(a0.v, b, acc[0][t]);
      acc[1][t] = WMMA_BF16(a1.v, b, acc[1][t]);
    }
    __syncthreads();
  }

#pragma unroll
  for (int g = 0; g < 2; ++g) {
    const int rbase = m_wave + g * 16 + ((lane >> 4) << 3);
#pragma unroll
    for (int t = 0; t < 4; ++t) {
      const int cg = n_base + 16 * t + (lane & 15);
      const float bv = bias[cg];
#pragma unroll
      for (int v = 0; v < 8; ++v) {
        const size_t idx = (size_t)(rbase + v) * HID + cg;
        hout[idx] = (__bf16)tanhf(acc[g][t][v] + xproj[idx] + bv);
      }
    }
  }
}

// ---------------------------------------------------------------------------
// (3) LM head: logits = h @ lm_head^T + b   (f32 out, 8192 x 32000)
//     Same LDS-staged 256x64 block structure.
// ---------------------------------------------------------------------------
__global__ __launch_bounds__(256) void k_lmhead(
    const __bf16* __restrict__ hin,      // [M_TOTAL, HID] bf16
    const __bf16* __restrict__ Wlm,      // [VOCAB, HID] bf16
    const float*  __restrict__ bias,     // [VOCAB]
    float*        __restrict__ out)      // [M_TOTAL, VOCAB]
{
  __shared__ __bf16 Bs[2][64 * 32];

  const int tid     = threadIdx.x;
  const int lane    = tid & 31;
  const int wave    = tid >> 5;
  const int ngroups = VOCABSZ / 64;              // 500
  const int m_blk   = (blockIdx.x / ngroups) * 256;
  const int n_base  = (blockIdx.x % ngroups) * 64;

  const int scol  = tid >> 2;
  const int skoff = (tid & 3) * 8;
  const __bf16* wsrc = Wlm + (size_t)(n_base + scol) * HID + skoff;
  const int sidx = scol * 32 + skoff;

  const int m_wave   = m_blk + wave * 32;
  const __bf16* ap0  = hin + (size_t)(m_wave + (lane & 15)) * HID;
  const __bf16* ap1  = ap0 + (size_t)16 * HID;
  const int koff     = (lane >> 4) << 3;
  const int kb_lane  = (lane >> 4) << 4;
  const int bcolbase = (lane & 15) * 32 + kb_lane;

  f32x8 acc[2][4] = {};

  *(bf16x8*)(&Bs[0][sidx]) = *(const bf16x8*)(wsrc);
  __syncthreads();

  const int NK = HID / 32;
  for (int kc = 0; kc < NK; ++kc) {
    const int cur = kc & 1;
    if (kc + 1 < NK)
      *(bf16x8*)(&Bs[cur ^ 1][sidx]) = *(const bf16x8*)(wsrc + (size_t)(kc + 1) * 32);

    const int k0 = kc * 32;
    AFrag a0, a1;
    a0.h8[0] = *(const bf16x8*)(ap0 + k0 + kb_lane);
    a0.h8[1] = *(const bf16x8*)(ap0 + k0 + kb_lane + 8);
    a1.h8[0] = *(const bf16x8*)(ap1 + k0 + kb_lane);
    a1.h8[1] = *(const bf16x8*)(ap1 + k0 + kb_lane + 8);
#pragma unroll
    for (int t = 0; t < 4; ++t) {
      bf16x16 b = *(const bf16x16*)(&Bs[cur][bcolbase + t * (16 * 32)]);
      acc[0][t] = WMMA_BF16(a0.v, b, acc[0][t]);
      acc[1][t] = WMMA_BF16(a1.v, b, acc[1][t]);
    }
    __syncthreads();
  }

#pragma unroll
  for (int g = 0; g < 2; ++g) {
    const int rbase = m_wave + g * 16 + ((lane >> 4) << 3);
#pragma unroll
    for (int t = 0; t < 4; ++t) {
      const int cg = n_base + 16 * t + (lane & 15);
      const float bv = bias[cg];
#pragma unroll
      for (int v = 0; v < 8; ++v)
        out[(size_t)(rbase + v) * VOCABSZ + cg] = acc[g][t][v] + bv;
    }
  }
}

// ------------------------------- launcher ----------------------------------

extern "C" void kernel_launch(void* const* d_in, const int* in_sizes, int n_in,
                              void* d_out, int out_size, void* d_ws, size_t ws_size,
                              hipStream_t stream) {
  const int*   x         = (const int*)  d_in[0];
  const float* token_emb = (const float*)d_in[1];
  const float* pos_emb   = (const float*)d_in[2];
  const float* W_in_w    = (const float*)d_in[3];
  const float* W_in_b    = (const float*)d_in[4];
  const float* W_rec_w   = (const float*)d_in[5];
  const float* W_rec_b   = (const float*)d_in[6];
  const float* lm_head_w = (const float*)d_in[7];
  const float* lm_head_b = (const float*)d_in[8];
  float* out = (float*)d_out;

  // workspace carve-up (256B aligned)
  char* ws = (char*)d_ws;
  size_t off = 0;
  auto carve = [&](size_t bytes) {
    void* p = ws + off;
    off = (off + bytes + 255) & ~(size_t)255;
    return p;
  };
  __bf16* wrec_bf = (__bf16*)carve((size_t)HID * HID * 2);
  __bf16* win_bf  = (__bf16*)carve((size_t)HID * HID * 2);
  __bf16* wlm_bf  = (__bf16*)carve((size_t)VOCABSZ * HID * 2);
  float*  xproj   = (float*) carve((size_t)M_TOTAL * HID * 4);
  __bf16* hA      = (__bf16*)carve((size_t)M_TOTAL * HID * 2);
  __bf16* hB      = (__bf16*)carve((size_t)M_TOTAL * HID * 2);

  // one-shot weight conversions + h0 = 0
  k_f32_to_bf16<<<1024, 256, 0, stream>>>(W_rec_w,   wrec_bf, HID * HID);
  k_f32_to_bf16<<<1024, 256, 0, stream>>>(W_in_w,    win_bf,  HID * HID);
  k_f32_to_bf16<<<4096, 256, 0, stream>>>(lm_head_w, wlm_bf,  VOCABSZ * HID);
  k_zero_bf16  <<<2048, 256, 0, stream>>>(hA, M_TOTAL * HID);

  // x_proj = embed @ W_in^T + b   (8192 waves = 1024 blocks)
  k_embed_inproj<<<(M_TOTAL / 16) * (HID / 64) / 8, 256, 0, stream>>>(
      x, token_emb, pos_emb, win_bf, W_in_b, xproj);

  // 20 recurrent steps, double-buffered bf16 h
  // grid: (8192/256) m-blocks * (1024/64) n-groups = 512 blocks
  __bf16* cur = hA;
  __bf16* nxt = hB;
  for (int s = 0; s < EQ_STEPS; ++s) {
    k_rnn_step<<<(M_TOTAL / 256) * (HID / 64), 256, 0, stream>>>(
        cur, wrec_bf, xproj, W_rec_b, nxt);
    __bf16* t = cur; cur = nxt; nxt = t;
  }

  // logits: (8192/256) * (32000/64) = 16000 blocks
  k_lmhead<<<(M_TOTAL / 256) * (VOCABSZ / 64), 256, 0, stream>>>(
      cur, wlm_bf, lm_head_b, out);
}
